// AttentionConv_83906481095339
// MI455X (gfx1250) — matrix operand order repaired and checked by reading, not compiled, not run on hardware
//
#include <hip/hip_runtime.h>
#include <hip/hip_bf16.h>
#include <math.h>

// Problem constants (B, C, H, W) = (4, 128, 64, 64)
#define BATCH 4
#define CCH   128
#define N_PIX 4096   // 64*64
#define LDA   40     // LDS row stride in halves (80 B: 16-B aligned frag loads, 5 uint4)
#define LDB   40

typedef __attribute__((ext_vector_type(8)))  _Float16 v8h;
typedef __attribute__((ext_vector_type(16))) _Float16 v16h;
typedef __attribute__((ext_vector_type(8)))  float    v8f;

// ---------------------------------------------------------------------------
// WMMA fragment assembly, per CDNA5 ISA 7.12.2 (wave32), vectorized:
// A (16x32 f16) row-major tile As[m][k], stride LDA:
//   lane -> M = lane&15, half = lane>>4; needs K[8h..8h+7] and K[16+8h..+7]
//   = two contiguous 16-byte LDS loads.
// B (32x16 f16) stored COLUMN-major Bt[n][k], stride LDB:
//   lane -> N = lane&15, half = lane>>4; needs K[8h..8h+7] and K[16+8h..+7]
//   = two contiguous 16-byte LDS loads.
// C/D (16x16 f32): vgpr r -> row = r + 8*half, col = lane&15
// ---------------------------------------------------------------------------
__device__ inline v16h load_a_frag16(const _Float16* __restrict__ As, int lane,
                                     int rowBase) {
  const int m = rowBase + (lane & 15);
  const int h = (lane >> 4) * 8;
  const v8h lo = *(const v8h*)(As + m * LDA + h);
  const v8h hi = *(const v8h*)(As + m * LDA + 16 + h);
  return __builtin_shufflevector(lo, hi, 0, 1, 2, 3, 4, 5, 6, 7,
                                 8, 9, 10, 11, 12, 13, 14, 15);
}

__device__ inline v16h load_b_frag16(const _Float16* __restrict__ Bt, int lane,
                                     int colBase) {
  const int n = colBase + (lane & 15);
  const int h = (lane >> 4) * 8;
  const v8h lo = *(const v8h*)(Bt + n * LDB + h);
  const v8h hi = *(const v8h*)(Bt + n * LDB + 16 + h);
  return __builtin_shufflevector(lo, hi, 0, 1, 2, 3, 4, 5, 6, 7,
                                 8, 9, 10, 11, 12, 13, 14, 15);
}

__device__ inline unsigned pack2f16(float a, float b) {
  const unsigned short ha = __builtin_bit_cast(unsigned short, (_Float16)a);
  const unsigned short hb = __builtin_bit_cast(unsigned short, (_Float16)b);
  return (unsigned)ha | ((unsigned)hb << 16);
}

// ---------------------------------------------------------------------------
// K1: f[b,n] = Wf . x[b,:,n] + bf ; g likewise.   grid = B*N/256
// ---------------------------------------------------------------------------
__global__ __launch_bounds__(256) void fg_kernel(
    const float* __restrict__ x,
    const float* __restrict__ Wf, const float* __restrict__ bf,
    const float* __restrict__ Wg, const float* __restrict__ bg,
    float* __restrict__ f, float* __restrict__ g) {
  __shared__ float wf_s[CCH], wg_s[CCH];
  const int tid = threadIdx.x;
  if (tid < CCH) { wf_s[tid] = Wf[tid]; wg_s[tid] = Wg[tid]; }
  __syncthreads();

  const int gid = blockIdx.x * 256 + tid;     // b*N + n
  const int b   = gid >> 12;
  const int n   = gid & (N_PIX - 1);
  const float* xp = x + (size_t)b * CCH * N_PIX + n;
  float sf = 0.f, sg = 0.f;
#pragma unroll 4
  for (int c = 0; c < CCH; ++c) {
    const float xv = xp[(size_t)c * N_PIX];
    sf = fmaf(wf_s[c], xv, sf);
    sg = fmaf(wg_s[c], xv, sg);
  }
  f[gid] = sf + bf[0];
  g[gid] = sg + bg[0];
}

// ---------------------------------------------------------------------------
// K2: per-batch max/min of g (closed-form row max of rank-1 logits). grid = B
// ---------------------------------------------------------------------------
__global__ __launch_bounds__(256) void gstat_kernel(
    const float* __restrict__ g, float* __restrict__ gstat) {
  __shared__ float smax[256], smin[256];
  const int b = blockIdx.x, tid = threadIdx.x;
  float mx = -INFINITY, mn = INFINITY;
  for (int i = tid; i < N_PIX; i += 256) {
    const float v = g[b * N_PIX + i];
    mx = fmaxf(mx, v);
    mn = fminf(mn, v);
  }
  smax[tid] = mx; smin[tid] = mn;
  __syncthreads();
  for (int s = 128; s > 0; s >>= 1) {
    if (tid < s) {
      smax[tid] = fmaxf(smax[tid], smax[tid + s]);
      smin[tid] = fminf(smin[tid], smin[tid + s]);
    }
    __syncthreads();
  }
  if (tid == 0) { gstat[2 * b] = smax[0]; gstat[2 * b + 1] = smin[0]; }
}

// ---------------------------------------------------------------------------
// K3: zinv[b,j] = 1 / sum_i exp(f_j * g_i - m_j).  grid = B*N/256
// ---------------------------------------------------------------------------
__global__ __launch_bounds__(256) void zrow_kernel(
    const float* __restrict__ f, const float* __restrict__ g,
    const float* __restrict__ gstat, float* __restrict__ zinv) {
  __shared__ float gs[N_PIX];                 // 16 KB
  const int tid = threadIdx.x;
  const int b   = blockIdx.x >> 4;            // 16 blocks per batch
  const int j   = (blockIdx.x & 15) * 256 + tid;
  for (int i = tid; i < N_PIX; i += 256) gs[i] = g[b * N_PIX + i];
  __syncthreads();

  const float fj = f[b * N_PIX + j];
  const float m  = (fj >= 0.f) ? fj * gstat[2 * b] : fj * gstat[2 * b + 1];
  float sum = 0.f;
#pragma unroll 4
  for (int i = 0; i < N_PIX; ++i) sum += __expf(fj * gs[i] - m);
  zinv[b * N_PIX + j] = 1.f / sum;
}

// ---------------------------------------------------------------------------
// K4: hq[b,c,j] = (f16)(Wh @ x + bh)   -- WMMA f16, 128x64 tile per block.
// grid = B * (N/64), block = 256 (8 waves; wave w owns rows [16w,16w+16)).
// ---------------------------------------------------------------------------
__global__ __launch_bounds__(256) void proj_h_kernel(
    const float* __restrict__ x, const float* __restrict__ Wh,
    const float* __restrict__ bh, _Float16* __restrict__ hq) {
  __shared__ __align__(16) _Float16 As[CCH * LDA];   // Wh tile (128 x 32)
  __shared__ __align__(16) _Float16 Bt[64 * LDB];    // x tile, column-major

  const int tid  = threadIdx.x;
  const int lane = tid & 31;
  const int wave = tid >> 5;
  const int b    = blockIdx.x >> 6;
  const int j0   = (blockIdx.x & 63) * 64;

  v8f acc[4];
#pragma unroll
  for (int t = 0; t < 4; ++t)
#pragma unroll
    for (int e = 0; e < 8; ++e) acc[t][e] = 0.f;

  unsigned* AsU = (unsigned*)As;
  for (int k0 = 0; k0 < CCH; k0 += 32) {
    __syncthreads();
    // Stage Wh rows as packed f16 pairs (2048 dwords / 256 threads)
#pragma unroll
    for (int it = 0; it < 8; ++it) {
      const int idx = tid + it * 256;
      const int r = idx >> 4, kk2 = idx & 15;
      const float w0 = Wh[r * CCH + k0 + 2 * kk2];
      const float w1 = Wh[r * CCH + k0 + 2 * kk2 + 1];
      AsU[r * (LDA / 2) + kk2] = pack2f16(w0, w1);
    }
    // Stage x tile column-major: coalesced global read, scalar LDS scatter
#pragma unroll
    for (int it = 0; it < 8; ++it) {
      const int idx = tid + it * 256;
      const int kk = idx >> 6, ii = idx & 63;
      Bt[ii * LDB + kk] =
          (_Float16)x[((size_t)b * CCH + (k0 + kk)) * N_PIX + j0 + ii];
    }
    __syncthreads();

    const v16h aF = load_a_frag16(As, lane, wave * 16);
    v16h bF[4];
#pragma unroll
    for (int t = 0; t < 4; ++t) bF[t] = load_b_frag16(Bt, lane, t * 16);
#pragma unroll
    for (int t = 0; t < 4; ++t)
      acc[t] = __builtin_amdgcn_wmma_f32_16x16x32_f16(
          false, aF, false, bF[t], (short)0, acc[t], false, false);
  }

  const int half = lane >> 4, col = lane & 15;
#pragma unroll
  for (int t = 0; t < 4; ++t)
#pragma unroll
    for (int r = 0; r < 8; ++r) {
      const int c = wave * 16 + r + 8 * half;
      const int j = j0 + t * 16 + col;
      hq[((size_t)b * CCH + c) * N_PIX + j] = (_Float16)(acc[t][r] + bh[c]);
    }
}

// ---------------------------------------------------------------------------
// K5 software-pipelined staging.
// issue:  global loads for tile k (A: 2 x b128 of hq, B: f/zinv scalars)
// store:  (after WMMAs) wait, generate E = exp(f_k*g_i - m_k)/Z_k, LDS stores
// ---------------------------------------------------------------------------
struct StageRegs {
  uint4 a0, a1;            // hq rows chunk (A tile)
  float f0, f1, z0, z1;    // per-thread f / zinv (kk2 = tid&15 is loop-invariant)
};

__device__ inline void stage_issue(StageRegs& r, const uint4* __restrict__ hq128,
                                   const float* __restrict__ f,
                                   const float* __restrict__ zinv,
                                   int b, int k0, int tid) {
  const int c0 = tid >> 2, q0 = tid & 3;            // 512 uint4 / 256 threads
  const int c1 = (tid + 256) >> 2, q1 = tid & 3;
  r.a0 = hq128[((((size_t)b * CCH + c0) * N_PIX + k0) >> 3) + q0];
  r.a1 = hq128[((((size_t)b * CCH + c1) * N_PIX + k0) >> 3) + q1];
  const int kk = (tid & 15) * 2;
  r.f0 = f[b * N_PIX + k0 + kk];
  r.f1 = f[b * N_PIX + k0 + kk + 1];
  r.z0 = zinv[b * N_PIX + k0 + kk];
  r.z1 = zinv[b * N_PIX + k0 + kk + 1];
}

__device__ inline void stage_store(const StageRegs& r, _Float16* __restrict__ AsB,
                                   _Float16* __restrict__ BtB,
                                   const float* gI_s, int tid,
                                   float gmax, float gmin) {
  uint4* AsV = (uint4*)AsB;                         // LDA = 5 uint4 per row
  const int c0 = tid >> 2, q0 = tid & 3;
  const int c1 = (tid + 256) >> 2;
  AsV[c0 * 5 + q0] = r.a0;
  AsV[c1 * 5 + q0] = r.a1;
  const float m0 = (r.f0 >= 0.f) ? r.f0 * gmax : r.f0 * gmin;
  const float m1 = (r.f1 >= 0.f) ? r.f1 * gmax : r.f1 * gmin;
  unsigned* BtU = (unsigned*)BtB;
  const int kk2 = tid & 15;
#pragma unroll
  for (int it = 0; it < 4; ++it) {                  // 64 ii x 16 kk2 pairs
    const int ii = (tid + it * 256) >> 4;
    const float e0 = __expf(r.f0 * gI_s[ii] - m0) * r.z0;
    const float e1 = __expf(r.f1 * gI_s[ii] - m1) * r.z1;
    BtU[ii * (LDB / 2) + kk2] = pack2f16(e0, e1);
  }
}

// ---------------------------------------------------------------------------
// K5: out[b,c,i] = sum_j hq[b,c,j] * exp(f_j*g_i - m_j) * zinv_j + x[b,c,i]
// Pipelined: issue global loads (k+1) -> WMMAs (k) -> wait+store (k+1).
// grid = B * (N/64), block = 256.
// ---------------------------------------------------------------------------
__global__ __launch_bounds__(256) void attn_bmm_kernel(
    const float* __restrict__ x, const _Float16* __restrict__ hq,
    const float* __restrict__ f, const float* __restrict__ g,
    const float* __restrict__ zinv, const float* __restrict__ gstat,
    float* __restrict__ out) {
  __shared__ __align__(16) _Float16 As[2][CCH * LDA];  // 2 x 10240 B
  __shared__ __align__(16) _Float16 Bt[2][64 * LDB];   // 2 x  5120 B
  __shared__ float gI[64];

  const int tid  = threadIdx.x;
  const int lane = tid & 31;
  const int wave = tid >> 5;
  const int b    = blockIdx.x >> 6;
  const int i0   = (blockIdx.x & 63) * 64;

  const float gmax = gstat[2 * b], gmin = gstat[2 * b + 1];
  if (tid < 64) gI[tid] = g[b * N_PIX + i0 + tid];

  v8f acc[4];
#pragma unroll
  for (int t = 0; t < 4; ++t)
#pragma unroll
    for (int e = 0; e < 8; ++e) acc[t][e] = 0.f;

  const uint4* __restrict__ hq128 = (const uint4*)hq;

  __syncthreads();                                   // gI visible
  StageRegs r;
  stage_issue(r, hq128, f, zinv, b, 0, tid);
  stage_store(r, As[0], Bt[0], gI, tid, gmax, gmin);
  __syncthreads();                                   // tile 0 staged

  int cur = 0;
  for (int k0 = 0; k0 < N_PIX; k0 += 32) {
    const bool more = (k0 + 32 < N_PIX);
    if (more) {
      stage_issue(r, hq128, f, zinv, b, k0 + 32, tid);  // issue, no wait yet
      if (k0 + 64 < N_PIX) {                            // deepen the pipeline
        const int c0 = tid >> 2;
        __builtin_prefetch(
            &hq128[(((size_t)b * CCH + c0) * N_PIX + k0 + 64) >> 3], 0, 0);
      }
    }

    // Consume tile[cur]: load all fragments, then back-to-back WMMAs.
    const v16h aF = load_a_frag16(As[cur], lane, wave * 16);
    v16h bF[4];
#pragma unroll
    for (int t = 0; t < 4; ++t) bF[t] = load_b_frag16(Bt[cur], lane, t * 16);
#pragma unroll
    for (int t = 0; t < 4; ++t)
      acc[t] = __builtin_amdgcn_wmma_f32_16x16x32_f16(
          false, aF, false, bF[t], (short)0, acc[t], false, false);

    if (more)                                          // wait + exp + LDS store
      stage_store(r, As[cur ^ 1], Bt[cur ^ 1], gI, tid, gmax, gmin);
    __syncthreads();
    cur ^= 1;
  }

  // Residual add + store (C/D layout: row = r + 8*half, col = lane&15)
  const int half = lane >> 4, col = lane & 15;
#pragma unroll
  for (int t = 0; t < 4; ++t)
#pragma unroll
    for (int rr = 0; rr < 8; ++rr) {
      const int c = wave * 16 + rr + 8 * half;
      const int i = i0 + t * 16 + col;
      const size_t o = ((size_t)b * CCH + c) * N_PIX + i;
      out[o] = acc[t][rr] + x[o];
    }
}

// ---------------------------------------------------------------------------
// Launch
// ---------------------------------------------------------------------------
extern "C" void kernel_launch(void* const* d_in, const int* in_sizes, int n_in,
                              void* d_out, int out_size, void* d_ws, size_t ws_size,
                              hipStream_t stream) {
  const float* x  = (const float*)d_in[0];
  const float* Wf = (const float*)d_in[1];
  const float* bf = (const float*)d_in[2];
  const float* Wg = (const float*)d_in[3];
  const float* bg = (const float*)d_in[4];
  const float* Wh = (const float*)d_in[5];
  const float* bh = (const float*)d_in[6];
  float* out = (float*)d_out;

  // Workspace layout (hq offset is 16-B aligned: 3*B*N*4 + 32 bytes)
  float* f     = (float*)d_ws;                   // B*N
  float* g     = f + BATCH * N_PIX;              // B*N
  float* zinv  = g + BATCH * N_PIX;              // B*N
  float* gstat = zinv + BATCH * N_PIX;           // B*2
  _Float16* hq = (_Float16*)(gstat + 2 * BATCH); // B*C*N f16 (4 MB)

  fg_kernel<<<BATCH * N_PIX / 256, 256, 0, stream>>>(x, Wf, bf, Wg, bg, f, g);
  gstat_kernel<<<BATCH, 256, 0, stream>>>(g, gstat);
  zrow_kernel<<<BATCH * N_PIX / 256, 256, 0, stream>>>(f, g, gstat, zinv);
  proj_h_kernel<<<BATCH * (N_PIX / 64), 256, 0, stream>>>(x, Wh, bh, hq);
  attn_bmm_kernel<<<BATCH * (N_PIX / 64), 256, 0, stream>>>(x, hq, f, g, zinv,
                                                            gstat, out);
}
